// HGCNNet_28991029248704
// MI455X (gfx1250) — matrix operand order repaired and checked by reading, not compiled, not run on hardware
//
#include <hip/hip_runtime.h>
#include <hip/hip_bf16.h>
#include <math.h>

// ---------------- types for WMMA fragments ----------------
typedef __bf16        v16bf __attribute__((ext_vector_type(16)));
typedef float         v8f   __attribute__((ext_vector_type(8)));
typedef unsigned int  u32x4 __attribute__((ext_vector_type(4)));

union Frag {
    v16bf v;
    u32x4 q[2];
};

__device__ __forceinline__ unsigned short f2bf(float f) {
    unsigned int u = __float_as_uint(f);
    u += 0x7FFFu + ((u >> 16) & 1u);   // round-to-nearest-even
    return (unsigned short)(u >> 16);
}
__device__ __forceinline__ float bf2f(unsigned short h) {
    return __uint_as_float(((unsigned int)h) << 16);
}

__device__ __forceinline__ v8f wmma_bf16(const Frag& a, const Frag& b, v8f c) {
    return __builtin_amdgcn_wmma_f32_16x16x32_bf16(
        false, a.v, false, b.v, (short)0, c, false, false);
}

#define NN 50000
#define EE 1600000
#define FF 512

// ---------------- graph preprocessing ----------------
__global__ void k_init(float* deg, int* rowcnt, int* fill, int n) {
    int i = blockIdx.x * blockDim.x + threadIdx.x;
    if (i < n) { deg[i] = 1.0f; rowcnt[i] = 0; fill[i] = 0; }
}

__global__ void k_count(const int* erow, const int* ecol, float* deg, int* rowcnt, int e) {
    int i = blockIdx.x * blockDim.x + threadIdx.x;
    if (i < e) {
        atomicAdd(&deg[ecol[i]], 1.0f);      // exact: integer-valued counts
        atomicAdd(&rowcnt[erow[i]], 1);
    }
}

__global__ void k_dinv(const float* deg, float* dinv, int n) {
    int i = blockIdx.x * blockDim.x + threadIdx.x;
    if (i < n) { float d = deg[i]; dinv[i] = d > 0.0f ? rsqrtf(d) : 0.0f; }
}

// -------- exclusive scan of rowcnt -> rowstart (1024-wide blocks) --------
__global__ void k_scanA(const int* in, int* out, int* blksum, int n) {
    __shared__ int sh[1024];
    int tid = threadIdx.x;
    int i = blockIdx.x * 1024 + tid;
    int v = (i < n) ? in[i] : 0;
    sh[tid] = v;
    __syncthreads();
    for (int d = 1; d < 1024; d <<= 1) {
        int t = (tid >= d) ? sh[tid - d] : 0;
        __syncthreads();
        sh[tid] += t;
        __syncthreads();
    }
    if (i < n) out[i] = sh[tid] - v;             // exclusive
    if (tid == 1023) blksum[blockIdx.x] = sh[tid];
}

__global__ void k_scanB(int* blksum, int* blkoff, int nb) {
    if (threadIdx.x == 0 && blockIdx.x == 0) {
        int run = 0;
        for (int b = 0; b < nb; ++b) { blkoff[b] = run; run += blksum[b]; }
    }
}

__global__ void k_scanC(int* out, const int* blkoff, int n) {
    int i = blockIdx.x * 1024 + threadIdx.x;
    if (i < n) out[i] += blkoff[blockIdx.x];
}

__global__ void k_scatter(const int* erow, const int* ecol, const float* dinv,
                          const int* rowstart, int* fill, int* colss, float* valss, int e) {
    int i = blockIdx.x * blockDim.x + threadIdx.x;
    if (i < e) {
        int r = erow[i], c = ecol[i];
        int pos = rowstart[r] + atomicAdd(&fill[r], 1);
        colss[pos] = c;
        valss[pos] = dinv[r] * dinv[c];
    }
}

// deterministic summation order: sort each row's (col,val) by col
__global__ void k_sortrows(int* colss, float* valss, const int* rs, const int* rc, int n) {
    int i = blockIdx.x * blockDim.x + threadIdx.x;
    if (i >= n) return;
    int s = rs[i], m = rc[i];
    for (int a = 1; a < m; ++a) {
        int c = colss[s + a]; float v = valss[s + a];
        int b = a - 1;
        while (b >= 0 && colss[s + b] > c) {
            colss[s + b + 1] = colss[s + b];
            valss[s + b + 1] = valss[s + b];
            --b;
        }
        colss[s + b + 1] = c; valss[s + b + 1] = v;
    }
}

// ---------------- conversions / weight prep ----------------
__global__ void k_f2bf(const float* in, unsigned short* out, int n) {
    int i = blockIdx.x * blockDim.x + threadIdx.x;
    if (i < n) out[i] = f2bf(in[i]);
}

// w1t[n*512 + k] = W1[k*512 + n]
__global__ void k_w1t(const float* W1, unsigned short* w1t) {
    int idx = blockIdx.x * blockDim.x + threadIdx.x;
    if (idx < 512 * 512) {
        int n = idx >> 9, k = idx & 511;
        w1t[idx] = f2bf(W1[k * 512 + n]);
    }
}

// folded W2, transposed, 40->48 zero pad: w2ft[n*1536 + p*512 + k]
__global__ void k_w2fold(const float* W2, unsigned short* w2ft) {
    int idx = blockIdx.x * blockDim.x + threadIdx.x;
    if (idx >= 48 * 1536) return;
    int n = idx / 1536, r = idx % 1536;
    int p = r >> 9, k = r & 511;
    float v = 0.0f;
    if (n < 40) {
        if (p == 0)      v = W2[k * 40 + n] + W2[(512 + k) * 40 + n];
        else if (p == 1) v = W2[(1024 + k) * 40 + n] + W2[(1536 + k) * 40 + n];
        else             v = W2[(2048 + k) * 40 + n];
    }
    w2ft[idx] = f2bf(v);
}

// ---------------- GEMM1: t = relu(x @ W1 + b1), bf16 in/out ----------------
// grid.x = 1563 (2 M-tiles per block), block 256 = 8 waves;
// wave w computes cols [w*64, w*64+64) for both M-tiles (B-frag reused 2x)
__global__ void k_gemm_relu(const unsigned short* __restrict__ A,   // x bf16 [M,512]
                            const unsigned short* __restrict__ BT,  // W1^T bf16 [512,512]
                            const float* __restrict__ bias,
                            unsigned short* __restrict__ Out) {
    int wave = threadIdx.x >> 5, lane = threadIdx.x & 31;
    int lhalf = lane >> 4, lmod = lane & 15;
    int mt0 = blockIdx.x * 2;
    int mt1 = mt0 + 1;
    bool has2 = (mt1 < (NN / 16));               // wave-uniform guard
    const unsigned short* arow0 = A + (size_t)(mt0 * 16 + lmod) * FF;
    const unsigned short* arow1 = A + (size_t)(mt1 * 16 + lmod) * FF;
    int ntile0 = wave * 4;

    v8f acc[2][4] = {};
    for (int kt = 0; kt < 16; ++kt) {
        int kb = kt * 32;
        Frag a0, a1;
        a0.q[0] = *reinterpret_cast<const u32x4*>(arow0 + kb + lhalf * 8);
        a0.q[1] = *reinterpret_cast<const u32x4*>(arow0 + kb + 16 + lhalf * 8);
        if (has2) {
            a1.q[0] = *reinterpret_cast<const u32x4*>(arow1 + kb + lhalf * 8);
            a1.q[1] = *reinterpret_cast<const u32x4*>(arow1 + kb + 16 + lhalf * 8);
        }
        #pragma unroll
        for (int t = 0; t < 4; ++t) {
            int ncol = (ntile0 + t) * 16 + lmod;
            const unsigned short* bp = BT + (size_t)ncol * 512 + kb + lhalf * 8;
            Frag b;
            b.q[0] = *reinterpret_cast<const u32x4*>(bp);
            b.q[1] = *reinterpret_cast<const u32x4*>(bp + 16);
            acc[0][t] = wmma_bf16(a0, b, acc[0][t]);
            if (has2) acc[1][t] = wmma_bf16(a1, b, acc[1][t]);
        }
    }
    #pragma unroll
    for (int t = 0; t < 4; ++t) {
        int n = (ntile0 + t) * 16 + lmod;
        float bv = bias[n];
        #pragma unroll
        for (int i = 0; i < 8; ++i) {
            int mr0 = mt0 * 16 + i + lhalf * 8;
            float v = acc[0][t][i] + bv;
            v = v > 0.0f ? v : 0.0f;
            Out[(size_t)mr0 * FF + n] = f2bf(v);
            if (has2) {
                int mr1 = mt1 * 16 + i + lhalf * 8;
                float w = acc[1][t][i] + bv;
                w = w > 0.0f ? w : 0.0f;
                Out[(size_t)mr1 * FF + n] = f2bf(w);
            }
        }
    }
}

// ---------------- SpMM: out = dinv^2 * in + sum_e val*in[col], bf16 ----------------
// one block (256 threads) per row; each thread owns 2 adjacent feature cols.
// (col,val) pairs staged through LDS: one coalesced load per chunk, then
// conflict-free DS broadcasts instead of 256 redundant VMEM loads per edge.
#define SPMM_CH 256
__global__ void k_spmm(const unsigned short* __restrict__ tin,
                       unsigned short* __restrict__ tout,
                       const int* __restrict__ colss, const float* __restrict__ valss,
                       const int* __restrict__ rs, const int* __restrict__ rc,
                       const float* __restrict__ dinv) {
    __shared__ int   shc[SPMM_CH];
    __shared__ float shv[SPMM_CH];
    int row = blockIdx.x;
    int tid = threadIdx.x;
    int c0 = tid * 2;
    float di = dinv[row];
    float self = di * di;
    unsigned int u = *reinterpret_cast<const unsigned int*>(tin + (size_t)row * FF + c0);
    float a0 = self * __uint_as_float(u << 16);
    float a1 = self * __uint_as_float(u & 0xFFFF0000u);
    int s = rs[row], e = s + rc[row];
    for (int j0 = s; j0 < e; j0 += SPMM_CH) {
        int nchunk = e - j0;
        if (nchunk > SPMM_CH) nchunk = SPMM_CH;
        __syncthreads();
        if (tid < nchunk) {
            shc[tid] = colss[j0 + tid];
            shv[tid] = valss[j0 + tid];
        }
        __syncthreads();
        for (int jj = 0; jj < nchunk; ++jj) {
            int   c = shc[jj];
            float v = shv[jj];
            if (jj + 4 < nchunk) {
                int cn = shc[jj + 4];
                __builtin_prefetch(tin + (size_t)cn * FF + c0, 0, 1);
            }
            unsigned int w = *reinterpret_cast<const unsigned int*>(tin + (size_t)c * FF + c0);
            a0 += v * __uint_as_float(w << 16);
            a1 += v * __uint_as_float(w & 0xFFFF0000u);
        }
    }
    unsigned int o = (unsigned int)f2bf(a0) | ((unsigned int)f2bf(a1) << 16);
    *reinterpret_cast<unsigned int*>(tout + (size_t)row * FF + c0) = o;
}

// ---------------- GEMM2: logits = t@W2a + s1@W2b + s2@W2c + b2 ----------------
// grid.x = 3125, block 96 = 3 waves; wave w does n-tile w (48 padded cols)
__global__ void k_gemm2(const unsigned short* __restrict__ T,
                        const unsigned short* __restrict__ S1,
                        const unsigned short* __restrict__ S2,
                        const unsigned short* __restrict__ W2T,  // [48][1536]
                        const float* __restrict__ b2,
                        float* __restrict__ out) {
    int wave = threadIdx.x >> 5, lane = threadIdx.x & 31;
    int lhalf = lane >> 4, lmod = lane & 15;
    int mtile = blockIdx.x;
    int m = mtile * 16 + lmod;
    int ntile = wave;
    int ncol = ntile * 16 + lmod;
    const unsigned short* Aps[3] = {T, S1, S2};

    v8f acc = {};
    for (int p = 0; p < 3; ++p) {
        const unsigned short* arow = Aps[p] + (size_t)m * FF;
        for (int kt = 0; kt < 16; ++kt) {
            int kb = kt * 32;
            Frag a;
            a.q[0] = *reinterpret_cast<const u32x4*>(arow + kb + lhalf * 8);
            a.q[1] = *reinterpret_cast<const u32x4*>(arow + kb + 16 + lhalf * 8);
            const unsigned short* bp = W2T + (size_t)ncol * 1536 + p * 512 + kb + lhalf * 8;
            Frag b;
            b.q[0] = *reinterpret_cast<const u32x4*>(bp);
            b.q[1] = *reinterpret_cast<const u32x4*>(bp + 16);
            acc = wmma_bf16(a, b, acc);
        }
    }
    if (ncol < 40) {
        float bv = b2[ncol];
        #pragma unroll
        for (int i = 0; i < 8; ++i) {
            int mr = mtile * 16 + i + lhalf * 8;
            out[(size_t)mr * 40 + ncol] = acc[i] + bv;
        }
    }
}

// ---------------- log_softmax in place over 40 classes, wave per row ----------------
__global__ void k_logsoftmax(float* out, int n) {
    int wave = threadIdx.x >> 5, lane = threadIdx.x & 31;
    int row = blockIdx.x * 8 + wave;
    if (row >= n) return;
    float* p = out + (size_t)row * 40;
    float v0 = p[lane];
    float v1 = (lane < 8) ? p[32 + lane] : -3.4e38f;
    float mx = fmaxf(v0, v1);
    for (int o = 16; o > 0; o >>= 1) mx = fmaxf(mx, __shfl_xor(mx, o, 32));
    float s = expf(v0 - mx) + ((lane < 8) ? expf(v1 - mx) : 0.0f);
    for (int o = 16; o > 0; o >>= 1) s += __shfl_xor(s, o, 32);
    float l = mx + logf(s);
    p[lane] = v0 - l;
    if (lane < 8) p[32 + lane] = v1 - l;
}

// ---------------- host launcher ----------------
extern "C" void kernel_launch(void* const* d_in, const int* in_sizes, int n_in,
                              void* d_out, int out_size, void* d_ws, size_t ws_size,
                              hipStream_t stream) {
    (void)in_sizes; (void)n_in; (void)out_size; (void)ws_size;
    const float* x    = (const float*)d_in[0];
    const int*   erow = (const int*)d_in[1];
    const int*   ecol = (const int*)d_in[2];
    const float* W1   = (const float*)d_in[3];
    const float* b1   = (const float*)d_in[4];
    const float* W2   = (const float*)d_in[5];
    const float* b2   = (const float*)d_in[6];
    float* out = (float*)d_out;

    const int N = NN, E = EE, F = FF;

    char* base = (char*)d_ws;
    size_t off = 0;
    auto alloc = [&](size_t bytes) -> void* {
        void* p = base + off;
        off += (bytes + 255) & ~(size_t)255;
        return p;
    };
    float* deg      = (float*)alloc((size_t)N * 4);
    float* dinv     = (float*)alloc((size_t)N * 4);
    int*   rowcnt   = (int*)alloc((size_t)N * 4);
    int*   rowstart = (int*)alloc((size_t)N * 4);
    int*   fill     = (int*)alloc((size_t)N * 4);
    int*   blksum   = (int*)alloc(64 * 4);
    int*   blkoff   = (int*)alloc(64 * 4);
    int*   colss    = (int*)alloc((size_t)E * 4);
    float* valss    = (float*)alloc((size_t)E * 4);
    unsigned short* xbf  = (unsigned short*)alloc((size_t)N * F * 2);
    unsigned short* w1t  = (unsigned short*)alloc((size_t)512 * 512 * 2);
    unsigned short* w2ft = (unsigned short*)alloc((size_t)48 * 1536 * 2);
    unsigned short* tbf  = (unsigned short*)alloc((size_t)N * F * 2);
    unsigned short* s1   = (unsigned short*)alloc((size_t)N * F * 2);
    unsigned short* s2   = (unsigned short*)alloc((size_t)N * F * 2);

    const int NB = (N + 1023) / 1024;   // 49

    // graph normalization + CSR
    k_init<<<(N + 255) / 256, 256, 0, stream>>>(deg, rowcnt, fill, N);
    k_count<<<(E + 255) / 256, 256, 0, stream>>>(erow, ecol, deg, rowcnt, E);
    k_dinv<<<(N + 255) / 256, 256, 0, stream>>>(deg, dinv, N);
    k_scanA<<<NB, 1024, 0, stream>>>(rowcnt, rowstart, blksum, N);
    k_scanB<<<1, 32, 0, stream>>>(blksum, blkoff, NB);
    k_scanC<<<NB, 1024, 0, stream>>>(rowstart, blkoff, N);
    k_scatter<<<(E + 255) / 256, 256, 0, stream>>>(erow, ecol, dinv, rowstart, fill, colss, valss, E);
    k_sortrows<<<(N + 255) / 256, 256, 0, stream>>>(colss, valss, rowstart, rowcnt, N);

    // weight / input prep
    k_f2bf<<<((N * F) + 255) / 256, 256, 0, stream>>>(x, xbf, N * F);
    k_w1t<<<(512 * 512) / 256, 256, 0, stream>>>(W1, w1t);
    k_w2fold<<<(48 * 1536 + 255) / 256, 256, 0, stream>>>(W2, w2ft);

    // t = relu(x @ W1 + b1)
    k_gemm_relu<<<(N / 16 + 1) / 2, 256, 0, stream>>>(xbf, w1t, b1, tbf);

    // s1 = spmm(t), s2 = spmm(s1)
    k_spmm<<<N, 256, 0, stream>>>(tbf, s1, colss, valss, rowstart, rowcnt, dinv);
    k_spmm<<<N, 256, 0, stream>>>(s1, s2, colss, valss, rowstart, rowcnt, dinv);

    // logits + log_softmax
    k_gemm2<<<N / 16, 96, 0, stream>>>(tbf, s1, s2, w2ft, b2, out);
    k_logsoftmax<<<(N + 7) / 8, 256, 0, stream>>>(out, N);
}
